// GHMBCE_6545530159211
// MI455X (gfx1250) — compile-verified, matrix operand order checked by reading, not run on hardware
//
#include <hip/hip_runtime.h>
#include <hip/hip_bf16.h>
#include <math.h>

typedef __attribute__((ext_vector_type(8))) int v8i;

#define N_ELEMS 16384
#define DELTA_F 0.1f
#define EPS_F 1e-12f
#define STAGE 4096   // floats staged in LDS per outer iteration (16 KB)

// ---------------------------------------------------------------------------
// Kernel 1: elementwise g = |sigmoid(x) - t|, loss = stable BCE-with-logits
// ---------------------------------------------------------------------------
__global__ void ghm_prep(const float* __restrict__ logits,
                         const float* __restrict__ targets,
                         float* __restrict__ g,
                         float* __restrict__ loss,
                         int n) {
    int i = blockIdx.x * blockDim.x + threadIdx.x;
    if (i >= n) return;
    float x = logits[i];
    float t = targets[i];
    float pred = 1.0f / (1.0f + __expf(-x));
    g[i] = fabsf(pred - t);
    // max(x,0) - x*t + log1p(exp(-|x|))
    loss[i] = fmaxf(x, 0.0f) - x * t + log1pf(expf(-fabsf(x)));
}

// ---------------------------------------------------------------------------
// Build one 16x64 IU8 A-tile fragment for this lane.
// ISA layout (8-bit A, 16x64, wave32): lane L -> row M = L&15, halfsel = L>>4.
// VGPR r holds bytes for K = (r>>1)*16 + halfsel*8 + (r&1)*4 ... +3.
// Byte = 1 iff |g_j - g_i| <= DELTA.
// ---------------------------------------------------------------------------
__device__ __forceinline__ v8i build_a_tile(const float* __restrict__ sm,
                                            int jc, int halfsel, float gi) {
    v8i a;
#pragma unroll
    for (int r = 0; r < 8; ++r) {
        int kbase = ((r >> 1) << 4) + (halfsel << 3) + ((r & 1) << 2);
        const float* p = sm + jc + kbase;
        unsigned b = 0;
#pragma unroll
        for (int t = 0; t < 4; ++t) {
            b |= (fabsf(p[t] - gi) <= DELTA_F ? 1u : 0u) << (8 * t);
        }
        a[r] = (int)b;
    }
    return a;
}

// ---------------------------------------------------------------------------
// Kernel 2: pairwise window counts via V_WMMA_I32_16X16X64_IU8.
// Block = 256 threads = 8 waves; each wave owns 16 consecutive rows.
// g[] is staged through LDS in 16 KB chunks shared by all 8 waves.
// D = A(0/1 bytes) x B(ones)  ->  every column of D holds the row count.
// ---------------------------------------------------------------------------
__global__ void __launch_bounds__(256)
ghm_paircount(const float* __restrict__ g,
              const float* __restrict__ loss,
              float* __restrict__ contrib,
              int n) {
    __shared__ float smem[STAGE];

    const int tid     = threadIdx.x;
    const int lane    = tid & 31;
    const int wave    = tid >> 5;
    const int halfsel = lane >> 4;
    const int i_base  = blockIdx.x * 128 + wave * 16;

    // This lane's row value (row M = lane & 15 of the wave's 16-row tile)
    const float gi = g[i_base + (lane & 15)];

    // B = all-ones bytes (layout-independent since every element is 1)
    v8i bmat;
#pragma unroll
    for (int r = 0; r < 8; ++r) bmat[r] = 0x01010101;

    // Two integer accumulators to break the WMMA->WMMA RAW chain (exact adds)
    v8i acc0 = {};
    v8i acc1 = {};

    for (int js = 0; js < n; js += STAGE) {
        __syncthreads();
#pragma unroll
        for (int k = 0; k < STAGE / 256; ++k) {
            smem[tid + k * 256] = g[js + tid + k * 256];
        }
        __syncthreads();

        for (int jc = 0; jc < STAGE; jc += 128) {
            v8i a0 = build_a_tile(smem, jc, halfsel, gi);
            acc0 = __builtin_amdgcn_wmma_i32_16x16x64_iu8(
                false, a0, false, bmat, acc0, false, false);
            v8i a1 = build_a_tile(smem, jc + 64, halfsel, gi);
            acc1 = __builtin_amdgcn_wmma_i32_16x16x64_iu8(
                false, a1, false, bmat, acc1, false, false);
        }
    }

    // 32-bit C/D 16x16 layout: lanes 0-15 VGPR r = D[M=r][N=lane],
    // lanes 16-31 VGPR r = D[M=8+r][N=lane-16]. All columns identical
    // (B = ones), so lane 0 carries rows 0..7 and lane 16 carries rows 8..15.
    if ((lane & 15) == 0) {
        const int rbase = i_base + halfsel * 8;
#pragma unroll
        for (int r = 0; r < 8; ++r) {
            float cnt = (float)(acc0[r] + acc1[r]);
            float GD  = cnt * 10.0f;                 // count / DELTA
            // (beta*loss).mean() with beta = n/(GD+eps): the n cancels.
            contrib[rbase + r] = loss[rbase + r] / (GD + EPS_F);
        }
    }
}

// ---------------------------------------------------------------------------
// Kernel 3: deterministic single-block reduction of contrib[] -> out[0]
// ---------------------------------------------------------------------------
__global__ void __launch_bounds__(256)
ghm_reduce(const float* __restrict__ contrib, float* __restrict__ out, int n) {
    __shared__ float sred[256];
    int tid = threadIdx.x;
    float s = 0.0f;
    for (int k = tid; k < n; k += 256) s += contrib[k];
    sred[tid] = s;
    __syncthreads();
#pragma unroll
    for (int off = 128; off > 0; off >>= 1) {
        if (tid < off) sred[tid] += sred[tid + off];
        __syncthreads();
    }
    if (tid == 0) out[0] = sred[0];
}

// ---------------------------------------------------------------------------
extern "C" void kernel_launch(void* const* d_in, const int* in_sizes, int n_in,
                              void* d_out, int out_size, void* d_ws, size_t ws_size,
                              hipStream_t stream) {
    const float* logits  = (const float*)d_in[0];
    const float* targets = (const float*)d_in[1];
    float* out = (float*)d_out;
    const int n = in_sizes[0];   // 16384

    float* g       = (float*)d_ws;           // n floats
    float* loss    = g + n;                  // n floats
    float* contrib = loss + n;               // n floats

    ghm_prep<<<(n + 255) / 256, 256, 0, stream>>>(logits, targets, g, loss, n);
    // 8 waves/block, 16 rows/wave -> 128 rows/block
    ghm_paircount<<<n / 128, 256, 0, stream>>>(g, loss, contrib, n);
    ghm_reduce<<<1, 256, 0, stream>>>(contrib, out, n);
}
